// VanillaRNN_46059229282554
// MI455X (gfx1250) — matrix operand order repaired and checked by reading, not compiled, not run on hardware
//
#include <hip/hip_runtime.h>
#include <hip/hip_bf16.h>
#include <math.h>

typedef __bf16 bf16;
typedef __attribute__((ext_vector_type(8)))  __bf16 v8bf;
typedef __attribute__((ext_vector_type(16))) __bf16 v16bf;
typedef __attribute__((ext_vector_type(8)))  float  v8f;
typedef __attribute__((ext_vector_type(4)))  unsigned tdm_g0_t;
typedef __attribute__((ext_vector_type(8)))  unsigned tdm_g1_t;

#define VOCAB 32000
#define EMB   1024
#define HID   1024
#define BB    16
#define TT    512
#define BT    (BB*TT)   /* 8192 */
#define NB    16        /* blocks in persistent scan kernel */

// ---- workspace byte offsets (all 256B aligned) ----
#define OFF_AEMB  0ull                    // bf16 [8192][1024]   16,777,216 B
#define OFF_WIH   16777216ull             // bf16 [1024][1024]    2,097,152 B
#define OFF_WHH   18874368ull             // bf16 [1024][1024]    2,097,152 B
#define OFF_WFC   20971520ull             // bf16 [32000][1024]  65,536,000 B
#define OFF_XP    86507520ull             // f32  [8192][1024]   33,554,432 B
#define OFF_HS    120061952ull            // bf16 [8192][1024]   16,777,216 B
#define OFF_HBUF  136839168ull            // bf16 [2][16][1024]      65,536 B
#define OFF_BAR   136904704ull            // u32 barrier counter

__device__ __forceinline__ v16bf cat8(v8bf a, v8bf b) {
  return __builtin_shufflevector(a, b, 0,1,2,3,4,5,6,7,8,9,10,11,12,13,14,15);
}

// ---- Tensor Data Mover: async 2D bf16 tile load Global -> LDS ---------------
// Builds a D# (groups 0+1) per CDNA5 ISA and issues
// tensor_load_to_lds (2-group form; VADDR2/3 = NULL). Tracked by TENSORcnt.
__device__ __forceinline__ void tdm_load_2d_bf16(unsigned lds_off, const void* gaddr,
                                                 unsigned tile_d0, unsigned tile_d1,
                                                 unsigned tensor_d0, unsigned tensor_d1,
                                                 unsigned stride_d0) {
  unsigned long long ga = (unsigned long long)gaddr;
  tdm_g0_t g0;
  g0[0] = 1u;                                   // count=1, user descriptor
  g0[1] = lds_off;                              // lds_addr (bytes)
  g0[2] = (unsigned)ga;                         // global_addr[31:0]
  g0[3] = (unsigned)(ga >> 32) | (2u << 30);    // global_addr[56:32] | type=2
  tdm_g1_t g1;
  g1[0] = 1u << 16;                             // data_size=1 (2 bytes/elem)
  g1[1] = (tensor_d0 & 0xFFFFu) << 16;          // tensor_dim0[15:0]
  g1[2] = (tensor_d0 >> 16) | ((tensor_d1 & 0xFFFFu) << 16);
  g1[3] = (tensor_d1 >> 16) | (tile_d0 << 16);  // tensor_dim1[31:16] | tile_dim0
  g1[4] = tile_d1;                              // tile_dim1 ([15:0]); tile_dim2=0
  g1[5] = stride_d0;                            // tensor_dim0_stride[31:0]
  g1[6] = 0;                                    // stride[47:32] | dim1_stride lo
  g1[7] = 0;
  asm volatile("tensor_load_to_lds %0, %1" :: "s"(g0), "s"(g1) : "memory");
}

// ---------------- f32 -> bf16 weight conversion (8 elems/thread) -------------
__global__ void f32_to_bf16_k(const float* __restrict__ src, bf16* __restrict__ dst,
                              size_t n8) {
  size_t i = (size_t)blockIdx.x * blockDim.x + threadIdx.x;
  if (i >= n8) return;
  const float4* s = (const float4*)(src + i * 8);
  float4 f0 = s[0], f1 = s[1];
  v8bf o;
  o[0] = (bf16)f0.x; o[1] = (bf16)f0.y; o[2] = (bf16)f0.z; o[3] = (bf16)f0.w;
  o[4] = (bf16)f1.x; o[5] = (bf16)f1.y; o[6] = (bf16)f1.z; o[7] = (bf16)f1.w;
  *((v8bf*)dst + i) = o;
}

// ---------------- embedding gather + cast to bf16 ----------------------------
__global__ void embed_gather_k(const int* __restrict__ x, const float* __restrict__ emb,
                               bf16* __restrict__ A) {
  size_t i   = (size_t)blockIdx.x * blockDim.x + threadIdx.x; // 8-elem group
  size_t row = i >> 7;             // EMB/8 = 128 groups per row
  size_t c8  = (i & 127) << 3;
  int tok = x[row];
  const float* s = emb + (size_t)tok * EMB + c8;
  float4 f0 = *(const float4*)s;
  float4 f1 = *(const float4*)(s + 4);
  v8bf o;
  o[0] = (bf16)f0.x; o[1] = (bf16)f0.y; o[2] = (bf16)f0.z; o[3] = (bf16)f0.w;
  o[4] = (bf16)f1.x; o[5] = (bf16)f1.y; o[6] = (bf16)f1.z; o[7] = (bf16)f1.w;
  *(v8bf*)(A + row * EMB + c8) = o;
}

// ---------------- bf16 WMMA GEMM: C[M,N] = A[M,K] * Bw[N,K]^T + bias ---------
// block tile 128x128, 256 threads = 8 waves in 2(m) x 4(n), wave tile 64x32.
// TDM double-buffered LDS staging: wave 0 issues tensor_load_to_lds for stage
// kb+1 while all waves compute stage kb from the other buffer.
__global__ void __launch_bounds__(256)
gemm_bf16_wmma_k(const bf16* __restrict__ A, const bf16* __restrict__ Bw,
                 float* __restrict__ C, const float* __restrict__ bias1,
                 const float* __restrict__ bias2, int M, int N, int K) {
  __shared__ bf16 As[2][128 * 32];
  __shared__ bf16 Bs[2][128 * 32];
  const int tid  = threadIdx.x;
  const int wid  = tid >> 5;
  const int lane = tid & 31;
  const int lh   = lane & 15;
  const int half = lane >> 4;
  const int wm   = wid >> 2;  // 0..1
  const int wn   = wid & 3;   // 0..3
  const size_t rowA0 = (size_t)blockIdx.y * 128;
  const size_t rowB0 = (size_t)blockIdx.x * 128;

  const unsigned asOff0 = (unsigned)(unsigned long long)(void*)&As[0][0];
  const unsigned asOff1 = (unsigned)(unsigned long long)(void*)&As[1][0];
  const unsigned bsOff0 = (unsigned)(unsigned long long)(void*)&Bs[0][0];
  const unsigned bsOff1 = (unsigned)(unsigned long long)(void*)&Bs[1][0];

  v8f acc[4][2];
#pragma unroll
  for (int mt = 0; mt < 4; ++mt)
#pragma unroll
    for (int nt = 0; nt < 2; ++nt) {
      v8f z = {0.f, 0.f, 0.f, 0.f, 0.f, 0.f, 0.f, 0.f};
      acc[mt][nt] = z;
    }

  const int kTiles = K >> 5;
  // prologue: stage 0 into buffer 0
  if (wid == 0) {
    tdm_load_2d_bf16(asOff0, A + rowA0 * (size_t)K, 32, 128, (unsigned)K, (unsigned)M,
                     (unsigned)K);
    tdm_load_2d_bf16(bsOff0, Bw + rowB0 * (size_t)K, 32, 128, (unsigned)K, (unsigned)N,
                     (unsigned)K);
  }

  for (int kb = 0; kb < kTiles; ++kb) {
    const int buf = kb & 1;
    if (kb + 1 < kTiles) {
      if (wid == 0) {   // issue next stage into other buffer (wave-level DMA)
        tdm_load_2d_bf16(buf ? asOff0 : asOff1,
                         A + rowA0 * (size_t)K + (size_t)(kb + 1) * 32,
                         32, 128, (unsigned)K, (unsigned)M, (unsigned)K);
        tdm_load_2d_bf16(buf ? bsOff0 : bsOff1,
                         Bw + rowB0 * (size_t)K + (size_t)(kb + 1) * 32,
                         32, 128, (unsigned)K, (unsigned)N, (unsigned)K);
      }
      __builtin_amdgcn_s_wait_tensorcnt(2);  // oldest pair (stage kb) retired
    } else {
      __builtin_amdgcn_s_wait_tensorcnt(0);
    }
    if (kb + 2 < kTiles) {  // warm L2 for the panel after next
      int r = tid >> 1;
      __builtin_prefetch(A + (rowA0 + r) * (size_t)K + (size_t)(kb + 2) * 32, 0, 1);
      __builtin_prefetch(Bw + (rowB0 + r) * (size_t)K + (size_t)(kb + 2) * 32, 0, 1);
    }
    __syncthreads();                         // stage kb visible to all waves

    const bf16* Ab = &As[buf][0];
    const bf16* Bb = &Bs[buf][0];
    v16bf Af[4], Bf[2];
#pragma unroll
    for (int mt = 0; mt < 4; ++mt) {
      int r = wm * 64 + mt * 16 + lh;
      v8bf a0 = *(const v8bf*)(Ab + r * 32 + half * 8);
      v8bf a1 = *(const v8bf*)(Ab + r * 32 + 16 + half * 8);
      Af[mt] = cat8(a0, a1);
    }
#pragma unroll
    for (int nt = 0; nt < 2; ++nt) {
      int r = wn * 32 + nt * 16 + lh;
      v8bf b0 = *(const v8bf*)(Bb + r * 32 + half * 8);
      v8bf b1 = *(const v8bf*)(Bb + r * 32 + 16 + half * 8);
      Bf[nt] = cat8(b0, b1);
    }
#pragma unroll
    for (int mt = 0; mt < 4; ++mt)
#pragma unroll
      for (int nt = 0; nt < 2; ++nt)
        acc[mt][nt] = __builtin_amdgcn_wmma_f32_16x16x32_bf16(
            false, Af[mt], false, Bf[nt], (short)0, acc[mt][nt], false, false);
    __syncthreads();  // all waves done reading buf before TDM overwrites it
  }

  // epilogue: D layout -> row = base + r + 8*half, col = base + (lane&15)
#pragma unroll
  for (int mt = 0; mt < 4; ++mt) {
#pragma unroll
    for (int nt = 0; nt < 2; ++nt) {
      size_t col = rowB0 + wn * 32 + nt * 16 + lh;
      float bsum = bias1 ? bias1[col] : 0.f;
      if (bias2) bsum += bias2[col];
#pragma unroll
      for (int r = 0; r < 8; ++r) {
        size_t row = rowA0 + wm * 64 + mt * 16 + r + half * 8;
        C[row * (size_t)N + col] = acc[mt][nt][r] + bsum;
      }
    }
  }
}

// ---------------- persistent recurrent scan ----------------------------------
// 16 blocks x 128 threads (4 waves). Block pins 64 columns of W_hh (128 KB LDS,
// loaded once via TDM). Wave w computes h_next[:, bid*64 + w*16 .. +16] via 32
// chained WMMAs per step; steps separated by an atomic inter-block barrier.
__global__ void __launch_bounds__(128)
rnn_scan_k(const float* __restrict__ xp, const bf16* __restrict__ Whh,
           bf16* __restrict__ hbuf, bf16* __restrict__ hs,
           float* __restrict__ hT, unsigned* __restrict__ bar) {
  extern __shared__ bf16 Ws[];  // [64][1024] bf16 = 128 KB
  const int tid  = threadIdx.x;
  const int bid  = blockIdx.x;
  const int wid  = tid >> 5;
  const int lane = tid & 31;
  const int lh   = lane & 15;
  const int half = lane >> 4;

  // one-shot TDM preload of this block's W_hh slice (64 rows x 1024 bf16)
  if (wid == 0) {
    tdm_load_2d_bf16((unsigned)(unsigned long long)(void*)&Ws[0],
                     Whh + (size_t)bid * 64 * HID,
                     1024, 64, 1024, 64, 1024);
  }
  __builtin_amdgcn_s_wait_tensorcnt(0);
  __syncthreads();

  const int nloc = wid * 16 + lh;           // local column (B-fragment lane col)
  const int ncol = bid * 64 + nloc;         // global column this lane owns in D

  for (int t = 0; t < TT; ++t) {
    const bf16* hin  = hbuf + (size_t)(t & 1) * (BB * HID);
    bf16*       hout = hbuf + (size_t)((t + 1) & 1) * (BB * HID);

    v8f acc = {0.f, 0.f, 0.f, 0.f, 0.f, 0.f, 0.f, 0.f};
#pragma unroll 4
    for (int kt = 0; kt < 32; ++kt) {
      const int k0 = kt * 32;
      // A fragment: h_old rows m=lh, K chunks per ISA 16-bit A layout
      v8bf a0 = *(const v8bf*)(hin + (size_t)lh * HID + k0 + half * 8);
      v8bf a1 = *(const v8bf*)(hin + (size_t)lh * HID + k0 + 16 + half * 8);
      v16bf Af = cat8(a0, a1);
      // B fragment from LDS: column nloc, K chunks half*8 and 16+half*8
      const bf16* wp = Ws + (size_t)nloc * HID + k0 + half * 8;
      v8bf b0 = *(const v8bf*)wp;
      v8bf b1 = *(const v8bf*)(wp + 16);
      v16bf Bf = cat8(b0, b1);
      acc = __builtin_amdgcn_wmma_f32_16x16x32_bf16(
          false, Af, false, Bf, (short)0, acc, false, false);
    }

    // epilogue: h = tanh(acc + xp[b, t, ncol]); rows b = r + 8*half
#pragma unroll
    for (int r = 0; r < 8; ++r) {
      int b = r + 8 * half;
      float v = tanhf(acc[r] + xp[((size_t)b * TT + t) * HID + ncol]);
      bf16 hv = (bf16)v;
      hout[(size_t)b * HID + ncol]                  = hv;  // next-step state
      hs[((size_t)b * TT + t) * HID + ncol]         = hv;  // sequence output
      if (t == TT - 1) hT[(size_t)b * HID + ncol]   = v;   // final hidden (f32)
    }

    // inter-block barrier (monotonic counter, s_sleep backoff)
    __threadfence();
    __syncthreads();
    if (tid == 0) {
      __hip_atomic_fetch_add(bar, 1u, __ATOMIC_ACQ_REL, __HIP_MEMORY_SCOPE_AGENT);
      const unsigned target = (unsigned)(t + 1) * NB;
      while (__hip_atomic_load(bar, __ATOMIC_ACQUIRE, __HIP_MEMORY_SCOPE_AGENT) < target)
        __builtin_amdgcn_s_sleep(1);
    }
    __syncthreads();
  }
}

// ---------------- launcher ---------------------------------------------------
extern "C" void kernel_launch(void* const* d_in, const int* in_sizes, int n_in,
                              void* d_out, int out_size, void* d_ws, size_t ws_size,
                              hipStream_t stream) {
  const int*   x    = (const int*)d_in[0];
  const float* emb  = (const float*)d_in[1];
  const float* Wih  = (const float*)d_in[2];
  const float* Whh  = (const float*)d_in[3];
  const float* bih  = (const float*)d_in[4];
  const float* bhh  = (const float*)d_in[5];
  const float* Wfc  = (const float*)d_in[6];
  const float* bfc  = (const float*)d_in[7];

  char* ws = (char*)d_ws;
  bf16*  Aemb  = (bf16*)(ws + OFF_AEMB);
  bf16*  wih_b = (bf16*)(ws + OFF_WIH);
  bf16*  whh_b = (bf16*)(ws + OFF_WHH);
  bf16*  wfc_b = (bf16*)(ws + OFF_WFC);
  float* xp    = (float*)(ws + OFF_XP);
  bf16*  hs    = (bf16*)(ws + OFF_HS);
  bf16*  hbuf  = (bf16*)(ws + OFF_HBUF);
  unsigned* bar = (unsigned*)(ws + OFF_BAR);

  float* logits  = (float*)d_out;                       // [8192][32000]
  float* hiddenT = logits + (size_t)BT * VOCAB;         // [16][1024]

  // 1) weight conversions to bf16
  f32_to_bf16_k<<<512,   256, 0, stream>>>(Wih, wih_b, (size_t)HID * EMB / 8);
  f32_to_bf16_k<<<512,   256, 0, stream>>>(Whh, whh_b, (size_t)HID * HID / 8);
  f32_to_bf16_k<<<16000, 256, 0, stream>>>(Wfc, wfc_b, (size_t)VOCAB * HID / 8);

  // 2) embedding gather -> bf16 A matrix
  embed_gather_k<<<4096, 256, 0, stream>>>(x, emb, Aemb);

  // 3) xp = Aemb * W_ih^T + b_ih + b_hh   (f32 out)
  {
    dim3 g(HID / 128, BT / 128);
    gemm_bf16_wmma_k<<<g, 256, 0, stream>>>(Aemb, wih_b, xp, bih, bhh, BT, HID, EMB);
  }

  // 4) zero h0 (both ping-pong buffers) and barrier counter
  hipMemsetAsync(ws + OFF_HBUF, 0, 2 * BB * HID * sizeof(bf16) + 256, stream);

  // 5) persistent recurrent scan (512 steps, inter-block barrier)
  rnn_scan_k<<<NB, 128, 64 * HID * sizeof(bf16), stream>>>(xp, whh_b, hbuf, hs,
                                                           hiddenT, bar);

  // 6) logits = hs * W_fc^T + b_fc
  {
    dim3 g(VOCAB / 128, BT / 128);
    gemm_bf16_wmma_k<<<g, 256, 0, stream>>>(hs, wfc_b, logits, bfc, nullptr,
                                            BT, VOCAB, HID);
  }
}
